// QuantLinear_25512105738706
// MI455X (gfx1250) — matrix operand order repaired and checked by reading, not compiled, not run on hardware
//
#include <hip/hip_runtime.h>
#include <hip/hip_bf16.h>

typedef __attribute__((ext_vector_type(16))) _Float16 v16h;
typedef __attribute__((ext_vector_type(8)))  _Float16 v8h;
typedef __attribute__((ext_vector_type(8)))  float    v8f;

// ---------------------------------------------------------------------------
// Problem dims (fixed by the reference)
// ---------------------------------------------------------------------------
static constexpr int SEQ   = 2048;
static constexpr int D_IN  = 11008;
static constexpr int D_OUT = 4096;

// ---------------------------------------------------------------------------
// GEMM tile config: C[M,N] = A[M,K] (f16, row-major) x B[N,K]^T (f16, row-major)
// f32 accumulate via v_wmma_f32_16x16x32_f16.  M,N multiples of 128, K of 32.
// ---------------------------------------------------------------------------
#define BM 128
#define BN 128
#define BK 32
#define LDSW 40          // padded LDS row stride in halfs (80B -> conflict-free b128 reads)

__device__ inline v16h frag_a(const _Float16* row, int fh) {
    // 16-bit A layout: lane half fh holds K = fh*8..fh*8+7 then 16+fh*8..16+fh*8+7
    v8h lo = *(const v8h*)(row + fh * 8);
    v8h hi = *(const v8h*)(row + 16 + fh * 8);
    return __builtin_shufflevector(lo, hi, 0,1,2,3,4,5,6,7,8,9,10,11,12,13,14,15);
}

__device__ inline v16h frag_b(const _Float16* row, int fh) {
    // 16-bit B layout: lane half fh holds K = fh*8..fh*8+7 then 16+fh*8..16+fh*8+7
    v8h lo = *(const v8h*)(row + fh * 8);
    v8h hi = *(const v8h*)(row + 16 + fh * 8);
    return __builtin_shufflevector(lo, hi, 0,1,2,3,4,5,6,7,8,9,10,11,12,13,14,15);
}

__global__ __launch_bounds__(256, 1)
void gemm_nt_wmma(const _Float16* __restrict__ A,   // [M,K]
                  const _Float16* __restrict__ B,   // [N,K]
                  float* __restrict__ C,            // [M,N]
                  const float* __restrict__ bias,   // [N] or nullptr
                  int M, int N, int K)
{
    __shared__ _Float16 As[BM * LDSW];
    __shared__ _Float16 Bs[BN * LDSW];

    const int tid  = threadIdx.x;
    const int wid  = tid >> 5;
    const int lane = tid & 31;
    const int fr   = lane & 15;   // fragment row (m or n within a 16x16 tile)
    const int fh   = lane >> 4;   // which half of the wave

    const int bm0 = blockIdx.y * BM;
    const int bn0 = blockIdx.x * BN;

    // 8 waves: 4 along M, 2 along N. Each wave computes 32(M) x 64(N).
    const int wm = (wid & 3) * 32;
    const int wn = (wid >> 2) * 64;

    v8f acc[2][4];
    #pragma unroll
    for (int i = 0; i < 2; ++i)
        #pragma unroll
        for (int j = 0; j < 4; ++j)
            acc[i][j] = v8f{};

    for (int k0 = 0; k0 < K; k0 += BK) {
        // ---- cooperative tile load: 512 chunks of 8 halfs each per matrix ----
        #pragma unroll
        for (int c = tid; c < (BM * BK / 8); c += 256) {
            const int row = c >> 2;
            const int col = (c & 3) * 8;
            *(v8h*)&As[row * LDSW + col] =
                *(const v8h*)(A + (size_t)(bm0 + row) * K + k0 + col);
            *(v8h*)&Bs[row * LDSW + col] =
                *(const v8h*)(B + (size_t)(bn0 + row) * K + k0 + col);
        }
        __syncthreads();

        // ---- fragments + 8 WMMAs ----
        v16h af[2], bf[4];
        #pragma unroll
        for (int mi = 0; mi < 2; ++mi)
            af[mi] = frag_a(&As[(wm + mi * 16 + fr) * LDSW], fh);
        #pragma unroll
        for (int ni = 0; ni < 4; ++ni)
            bf[ni] = frag_b(&Bs[(wn + ni * 16 + fr) * LDSW], fh);

        #pragma unroll
        for (int mi = 0; mi < 2; ++mi)
            #pragma unroll
            for (int ni = 0; ni < 4; ++ni)
                acc[mi][ni] = __builtin_amdgcn_wmma_f32_16x16x32_f16(
                    false, af[mi], false, bf[ni], (short)0, acc[mi][ni],
                    false, false);
        __syncthreads();
    }

    // ---- epilogue: C/D layout -> lane fr = N column, reg r -> M = r + 8*fh ----
    #pragma unroll
    for (int ni = 0; ni < 4; ++ni) {
        const int n = bn0 + wn + ni * 16 + fr;
        const float bv = bias ? bias[n] : 0.0f;
        #pragma unroll
        for (int mi = 0; mi < 2; ++mi) {
            #pragma unroll
            for (int r = 0; r < 8; ++r) {
                const int m = bm0 + wm + mi * 16 + fh * 8 + r;
                C[(size_t)m * N + n] = acc[mi][ni][r] + bv;
            }
        }
    }
}

// ---------------------------------------------------------------------------
// Tiled transpose with optional per-output-row scale:
//   dst[c][r] = src[r][c] * (scale ? scale[c] : 1), f32 -> f16
// ---------------------------------------------------------------------------
__global__ __launch_bounds__(256)
void transpose_scale_f32_f16(const float* __restrict__ src,   // [R, C]
                             _Float16* __restrict__ dst,      // [C, R]
                             const float* __restrict__ scale, // [C] or nullptr
                             int R, int C)
{
    __shared__ float tile[32][33];
    const int c0 = blockIdx.x * 32;
    const int r0 = blockIdx.y * 32;
    const int tx = threadIdx.x;          // 0..31
    const int ty = threadIdx.y;          // 0..7

    for (int i = ty; i < 32; i += 8)
        tile[i][tx] = src[(size_t)(r0 + i) * C + c0 + tx];
    __syncthreads();
    for (int i = ty; i < 32; i += 8) {
        const int c = c0 + i;
        const float s = scale ? scale[c] : 1.0f;
        dst[(size_t)c * R + r0 + tx] = (_Float16)(tile[tx][i] * s);
    }
}

// ---------------------------------------------------------------------------
// Elementwise f32 -> f16
// ---------------------------------------------------------------------------
__global__ __launch_bounds__(256)
void cvt_f32_f16(const float* __restrict__ src, _Float16* __restrict__ dst, size_t n)
{
    size_t i = (size_t)blockIdx.x * blockDim.x + threadIdx.x;
    const size_t stride = (size_t)gridDim.x * blockDim.x;
    for (; i < n; i += stride) dst[i] = (_Float16)src[i];
}

// ---------------------------------------------------------------------------
// brot[o] = sum_k Rt[o][k] * bias[k]   (Rt is the already-transposed f16 R)
// ---------------------------------------------------------------------------
__global__ __launch_bounds__(256)
void rot_bias(const _Float16* __restrict__ Rt, const float* __restrict__ bias,
              float* __restrict__ brot, int D)
{
    __shared__ float red[256];
    const int o = blockIdx.x;
    float s = 0.0f;
    for (int k = threadIdx.x; k < D; k += 256)
        s += (float)Rt[(size_t)o * D + k] * bias[k];
    red[threadIdx.x] = s;
    __syncthreads();
    for (int off = 128; off > 0; off >>= 1) {
        if (threadIdx.x < off) red[threadIdx.x] += red[threadIdx.x + off];
        __syncthreads();
    }
    if (threadIdx.x == 0) brot[o] = red[0];
}

// ---------------------------------------------------------------------------
// Per-row abs-max -> quant scale:  scale[r] = max(amax(row)/qmax, 1e-8)
// ---------------------------------------------------------------------------
__global__ __launch_bounds__(256)
void row_absmax_scale(const float* __restrict__ M, float* __restrict__ scale,
                      int cols, float qmax)
{
    __shared__ float red[256];
    const int r = blockIdx.x;
    float m = 0.0f;
    for (int c = threadIdx.x; c < cols; c += 256)
        m = fmaxf(m, fabsf(M[(size_t)r * cols + c]));
    red[threadIdx.x] = m;
    __syncthreads();
    for (int off = 128; off > 0; off >>= 1) {
        if (threadIdx.x < off)
            red[threadIdx.x] = fmaxf(red[threadIdx.x], red[threadIdx.x + off]);
        __syncthreads();
    }
    if (threadIdx.x == 0) scale[r] = fmaxf(red[0] / qmax, 1e-8f);
}

// ---------------------------------------------------------------------------
// 8-bit fake-quant of W -> f16 for the second GEMM (per-row scale)
// ---------------------------------------------------------------------------
__global__ __launch_bounds__(256)
void quant_rows_f16(const float* __restrict__ W, const float* __restrict__ scale,
                    _Float16* __restrict__ Wq, int cols, size_t n,
                    float lo, float hi)
{
    size_t i = (size_t)blockIdx.x * blockDim.x + threadIdx.x;
    const size_t stride = (size_t)gridDim.x * blockDim.x;
    for (; i < n; i += stride) {
        const float s = scale[i / cols];
        const float q = fminf(fmaxf(rintf(W[i] / s), lo), hi);
        Wq[i] = (_Float16)(q * s);
    }
}

// ---------------------------------------------------------------------------
// 16-bit fake-quant in place (per-token scale)
// ---------------------------------------------------------------------------
__global__ __launch_bounds__(256)
void quant_rows_inplace(float* __restrict__ M, const float* __restrict__ scale,
                        int cols, size_t n, float lo, float hi)
{
    size_t i = (size_t)blockIdx.x * blockDim.x + threadIdx.x;
    const size_t stride = (size_t)gridDim.x * blockDim.x;
    for (; i < n; i += stride) {
        const float s = scale[i / cols];
        const float q = fminf(fmaxf(rintf(M[i] / s), lo), hi);
        M[i] = q * s;
    }
}

// ---------------------------------------------------------------------------
// Host orchestration
// ---------------------------------------------------------------------------
extern "C" void kernel_launch(void* const* d_in, const int* in_sizes, int n_in,
                              void* d_out, int out_size, void* d_ws, size_t ws_size,
                              hipStream_t stream)
{
    const float* x      = (const float*)d_in[0];   // [SEQ, D_IN]
    const float* weight = (const float*)d_in[1];   // [D_OUT, D_IN]
    const float* bias   = (const float*)d_in[2];   // [D_OUT]
    const float* S_ud   = (const float*)d_in[3];   // [D_IN]
    const float* R      = (const float*)d_in[4];   // [D_OUT, D_OUT]
    float*       out    = (float*)d_out;           // [SEQ, D_OUT]

    // ---- carve workspace ----
    size_t off = 0;
    auto carve = [&](size_t bytes) -> void* {
        void* p = (char*)d_ws + off;
        off += (bytes + 255) & ~(size_t)255;
        return p;
    };
    _Float16* Rt   = (_Float16*)carve((size_t)D_OUT * D_OUT * sizeof(_Float16)); // R^T as [o][k]
    _Float16* Wst  = (_Float16*)carve((size_t)D_IN  * D_OUT * sizeof(_Float16)); // (weight*S)^T [i][k]; reused as Wq [o][i]
    float*    Wrot = (float*)   carve((size_t)D_OUT * D_IN  * sizeof(float));    // rotated W
    _Float16* x16  = (_Float16*)carve((size_t)SEQ   * D_IN  * sizeof(_Float16));
    float*    brot = (float*)   carve((size_t)D_OUT * sizeof(float));
    float*    wscl = (float*)   carve((size_t)D_OUT * sizeof(float));
    float*    oscl = (float*)   carve((size_t)SEQ   * sizeof(float));
    _Float16* Wq   = Wst;   // reuse after GEMM1 no longer needs Wst

    const dim3 tblk(32, 8);

    // 1) Rt[o][k] = R[k][o]  (f16)
    transpose_scale_f32_f16<<<dim3(D_OUT / 32, D_OUT / 32), tblk, 0, stream>>>(
        R, Rt, nullptr, D_OUT, D_OUT);

    // 2) Wst[i][k] = weight[k][i] * S_up_down[i]  (f16)
    transpose_scale_f32_f16<<<dim3(D_IN / 32, D_OUT / 32), tblk, 0, stream>>>(
        weight, Wst, S_ud, D_OUT, D_IN);

    // 3) x -> f16
    cvt_f32_f16<<<8192, 256, 0, stream>>>(x, x16, (size_t)SEQ * D_IN);

    // 4) Wrot[o][i] = sum_k Rt[o][k] * Wst[i][k]   (M=D_OUT, N=D_IN, K=D_OUT)
    gemm_nt_wmma<<<dim3(D_IN / BN, D_OUT / BM), 256, 0, stream>>>(
        Rt, Wst, Wrot, nullptr, D_OUT, D_IN, D_OUT);

    // 5) brot = R^T @ bias (via Rt)
    rot_bias<<<D_OUT, 256, 0, stream>>>(Rt, bias, brot, D_OUT);

    // 6) per-row 8-bit scales of Wrot
    row_absmax_scale<<<D_OUT, 256, 0, stream>>>(Wrot, wscl, D_IN, 127.0f);

    // 7) Wq[o][i] = fq8(Wrot) as f16  (K-contiguous for GEMM2)
    quant_rows_f16<<<16384, 256, 0, stream>>>(
        Wrot, wscl, Wq, D_IN, (size_t)D_OUT * D_IN, -128.0f, 127.0f);

    // 8) out[s][o] = sum_i x16[s][i] * Wq[o][i] + brot[o]  (M=SEQ, N=D_OUT, K=D_IN)
    gemm_nt_wmma<<<dim3(D_OUT / BN, SEQ / BM), 256, 0, stream>>>(
        x16, Wq, out, brot, SEQ, D_OUT, D_IN);

    // 9) per-token 16-bit scales
    row_absmax_scale<<<SEQ, 256, 0, stream>>>(out, oscl, D_OUT, 32767.0f);

    // 10) 16-bit fake-quant in place
    quant_rows_inplace<<<8192, 256, 0, stream>>>(
        out, oscl, D_OUT, (size_t)SEQ * D_OUT, -32768.0f, 32767.0f);
}
